// DownSample_64982855188998
// MI455X (gfx1250) — matrix operand order repaired and checked by reading, not compiled, not run on hardware
//
#include <hip/hip_runtime.h>

// Problem constants (from reference)
#define BB     128          // batch
#define T_IN   240
#define T_OUT  30
#define DD     128          // input feature dim
#define HH     256          // hidden
#define GG     768          // 3*H
#define M_ENC  (T_IN * BB)  // 30720 chain steps (encoder)
#define M_DEC  (T_OUT * BB) // 3840 chain steps (decoder)

typedef __attribute__((ext_vector_type(16))) __bf16 v16bf;
typedef __attribute__((ext_vector_type(8)))  float  v8f;

__device__ __forceinline__ void split_bf16(float v, __bf16& hi, __bf16& lo) {
  hi = (__bf16)v;                 // RNE truncation to bf16
  lo = (__bf16)(v - (float)hi);   // residual -> second bf16 (bf16x3 trick)
}

// ---------------------------------------------------------------------------
// Kernel 1: Gi[m][n] = sum_k X[row(m)][k] * W[n][k] + b[n]   (n in [0,768))
// row(m) maps chain-step m back into x: offset = (m%128)*T_IN*D + (m/128)*st
//   encoder: st = D (=128)      decoder: st = STRIDE*D (=1024)
// One wave per 16x16 output tile; K=128 consumed as 4 blocks of 32 with
// 3 bf16 WMMAs per block (hi*hi + hi*lo + lo*hi) for ~f32 accuracy.
// ---------------------------------------------------------------------------
__global__ __launch_bounds__(256)
void gi_gemm_wmma(const float* __restrict__ x,
                  const float* __restrict__ W,   // (768,128) row-major
                  const float* __restrict__ b,   // (768,)
                  float* __restrict__ Gi,        // (M,768)
                  int st)
{
  const int wave = threadIdx.x >> 5;
  const int lane = threadIdx.x & 31;
  const int hb   = lane >> 4;     // half-wave select
  const int lr   = lane & 15;

  const int tile = blockIdx.x * 8 + wave;   // grid sized exactly, no guard
  const int m0 = (tile / 48) * 16;
  const int n0 = (tile % 48) * 16;

  // A row for this lane (ISA: A rows live per-lane, M = lane%16)
  const int   m_a  = m0 + lr;
  const float* __restrict__ arow =
      x + (size_t)(m_a & (BB - 1)) * (T_IN * DD) + (size_t)(m_a >> 7) * st;
  // B column for this lane (ISA: B cols live per-lane, N = lane%16)
  const float* __restrict__ brow = W + (size_t)(n0 + lr) * DD;

  v8f acc = {};
#pragma unroll
  for (int k0 = 0; k0 < DD; k0 += 32) {
    v16bf ahi = {}, alo = {}, bhi = {}, blo = {};
#pragma unroll
    for (int j = 0; j < 16; ++j) {
      // 16-bit A 16x32 layout: lanes0-15 j<8->K=j, j>=8->K=j+8;
      //                        lanes16-31 j<8->K=j+8, j>=8->K=j+16
      int ka = k0 + ((j < 8) ? (j + (hb ? 8 : 0)) : (j + (hb ? 16 : 8)));
      float av = arow[ka];
      __bf16 h1, l1; split_bf16(av, h1, l1);
      ahi[j] = h1; alo[j] = l1;
      // B 32x16 layout: same K mapping as A (j<8->K=j+8*half, j>=8->K=j+8+8*half)
      int kb = k0 + ((j < 8) ? (j + (hb ? 8 : 0)) : (j + (hb ? 16 : 8)));
      float bv = brow[kb];
      __bf16 h2, l2; split_bf16(bv, h2, l2);
      bhi[j] = h2; blo[j] = l2;
    }
    acc = __builtin_amdgcn_wmma_f32_16x16x32_bf16(false, ahi, false, bhi,
                                                  (short)0, acc, false, false);
    acc = __builtin_amdgcn_wmma_f32_16x16x32_bf16(false, ahi, false, blo,
                                                  (short)0, acc, false, false);
    acc = __builtin_amdgcn_wmma_f32_16x16x32_bf16(false, alo, false, bhi,
                                                  (short)0, acc, false, false);
  }

  const float bias = b[n0 + lr];
  // C/D layout: VGPR r holds M = m0 + r + 8*half, N = n0 + lane%16
#pragma unroll
  for (int r = 0; r < 8; ++r) {
    int m = m0 + r + hb * 8;
    Gi[(size_t)m * GG + n0 + lr] = acc[r] + bias;
  }
}

// ---------------------------------------------------------------------------
// Kernel 2: the two sequential GRU chains (latency-bound part).
// block 0 = encoder (30720 steps), block 1 = decoder (3840 steps).
// 768 threads: phase A each thread computes one element of gh = h @ WhhT + bhh
// (h in LDS, Whh streamed from L2); phase B threads 0..255 apply the gates.
// ---------------------------------------------------------------------------
__device__ __forceinline__ float sigmoidf_(float v) {
  return 1.0f / (1.0f + __expf(-v));
}

__global__ __launch_bounds__(GG)
void gru_chains(const float* __restrict__ Gi_enc,
                const float* __restrict__ Whh_enc,
                const float* __restrict__ bhh_enc,
                const float* __restrict__ Gi_dec,
                const float* __restrict__ Whh_dec,
                const float* __restrict__ bhh_dec,
                float* __restrict__ out_enc,     // (240,256)
                float* __restrict__ out_dec)     // (128,30,256)
{
  __shared__ __align__(16) float h[HH];
  __shared__ float gh[GG];

  const int t = threadIdx.x;
  const float* Gi;  const float* Whh;  const float* bhh;
  int nsteps, mode;
  if (blockIdx.x == 0) { Gi = Gi_enc; Whh = Whh_enc; bhh = bhh_enc; nsteps = M_ENC; mode = 0; }
  else                 { Gi = Gi_dec; Whh = Whh_dec; bhh = bhh_dec; nsteps = M_DEC; mode = 1; }

  if (t < HH) h[t] = 0.0f;
  const float bb = bhh[t];
  const float* __restrict__ wr = Whh + (size_t)t * HH;
  __syncthreads();

  for (int s = 0; s < nsteps; ++s) {
    // --- phase A: gh[t] = dot(h, Whh[t,:]) + bhh[t] -----------------------
    if (s + 1 < nsteps)  // pull next step's Gi row toward L2/L0 while we work
      __builtin_prefetch(Gi + (size_t)(s + 1) * GG + t, 0, 0);
    float dot = 0.0f;
#pragma unroll 8
    for (int k = 0; k < HH; k += 4) {
      float4 w4 = *reinterpret_cast<const float4*>(wr + k);
      float4 h4 = *reinterpret_cast<const float4*>(h + k);   // ds_load_b128
      dot += w4.x * h4.x + w4.y * h4.y + w4.z * h4.z + w4.w * h4.w;
    }
    gh[t] = dot + bb;
    __syncthreads();

    // --- phase B: gates on threads 0..255 ---------------------------------
    if (t < HH) {
      const float* __restrict__ g = Gi + (size_t)s * GG;
      float r  = sigmoidf_(g[t]          + gh[t]);
      float z  = sigmoidf_(g[HH + t]     + gh[HH + t]);
      float n  = tanhf   (g[2 * HH + t]  + r * gh[2 * HH + t]);
      float hn = (1.0f - z) * n + z * h[t];
      h[t] = hn;
      if (mode == 0) {
        // encoder_hidden_states = hs_e[B-1::B] -> step s with s%128==127
        if ((s & (BB - 1)) == (BB - 1))
          out_enc[(size_t)(s >> 7) * HH + t] = hn;
      } else {
        // interped_seq[b][tt][:] with s = tt*128 + b
        out_dec[(size_t)(s & (BB - 1)) * (T_OUT * HH) +
                (size_t)(s >> 7) * HH + t] = hn;
      }
    }
    __syncthreads();
  }
}

// ---------------------------------------------------------------------------
extern "C" void kernel_launch(void* const* d_in, const int* in_sizes, int n_in,
                              void* d_out, int out_size, void* d_ws, size_t ws_size,
                              hipStream_t stream) {
  (void)in_sizes; (void)n_in; (void)out_size; (void)ws_size;
  const float* x       = (const float*)d_in[0];
  const float* Wih_enc = (const float*)d_in[1];
  const float* Whh_enc = (const float*)d_in[2];
  const float* bih_enc = (const float*)d_in[3];
  const float* bhh_enc = (const float*)d_in[4];
  const float* Wih_dec = (const float*)d_in[5];
  const float* Whh_dec = (const float*)d_in[6];
  const float* bih_dec = (const float*)d_in[7];
  const float* bhh_dec = (const float*)d_in[8];

  float* Gi_enc = (float*)d_ws;                       // 30720*768 f32
  float* Gi_dec = Gi_enc + (size_t)M_ENC * GG;        //  3840*768 f32

  float* out_enc = (float*)d_out;                     // (1,240,256)
  float* out_dec = out_enc + (size_t)T_IN * HH;       // (128,30,256)

  // Encoder Gi: 30720/16 * 768/16 = 92160 tile-waves -> 11520 blocks x 8 waves
  gi_gemm_wmma<<<11520, 256, 0, stream>>>(x, Wih_enc, bih_enc, Gi_enc, DD);
  // Decoder Gi (x[:, ::8, :]): 240 * 48 = 11520 tile-waves -> 1440 blocks
  gi_gemm_wmma<<<1440, 256, 0, stream>>>(x, Wih_dec, bih_dec, Gi_dec, 8 * DD);
  // Both sequential chains concurrently: block 0 = encoder, block 1 = decoder
  gru_chains<<<2, GG, 0, stream>>>(Gi_enc, Whh_enc, bhh_enc,
                                   Gi_dec, Whh_dec, bhh_dec,
                                   out_enc, out_dec);
}
